// DihedralsPredictor_11433202942437
// MI455X (gfx1250) — matrix operand order repaired and checked by reading, not compiled, not run on hardware
//
#include <hip/hip_runtime.h>
#include <hip/hip_bf16.h>
#include <math.h>

// ---------------- problem constants (match reference) ----------------
constexpr int NNODES = 30000;
constexpr int NEDGES = 480000;
constexpr int DMUL   = 128;   // feature dim
constexpr int NATTR  = 4;     // node_attr dim
constexpr int SHD    = 9;     // edge_attr dim
constexpr int NBAS   = 16;    // radial basis
constexpr int NLAY   = 4;     // conv layers
constexpr int NGR    = 8;     // graphs
constexpr int NDIH   = 10;
constexpr float INV_SQRT_Z = 0.25f;  // 1/sqrt(16)

typedef __attribute__((ext_vector_type(16))) _Float16 v16h;
typedef __attribute__((ext_vector_type(8)))  _Float16 v8h;
typedef __attribute__((ext_vector_type(8)))  float    v8f;

// ---------------- helpers ----------------
__device__ __forceinline__ float gelu_f(float x) {
  // tanh approximation (jax.nn.gelu default approximate=True)
  const float c = 0.7978845608028654f;
  float x3 = x * x * x;
  return 0.5f * x * (1.0f + tanhf(c * (x + 0.044715f * x3)));
}

__device__ __forceinline__ int lane_id() { return threadIdx.x & 31; }

__device__ __forceinline__ v8f wmma16x16x32(v16h a, v16h b, v8f c) {
  return __builtin_amdgcn_wmma_f32_16x16x32_f16(false, a, false, b, (short)0, c,
                                                false, false);
}

__device__ __forceinline__ v16h cat8(v8h lo, v8h hi) {
  return __builtin_shufflevector(lo, hi, 0, 1, 2, 3, 4, 5, 6, 7,
                                 8, 9, 10, 11, 12, 13, 14, 15);
}

// A fragment: 16x32 f16 tile from row-major f16 buffer (ld elems).
// Lane layout: lane m=lane&15 holds row m; its 16 halfs are the contiguous
// run k = half*16 .. half*16+15  -> 2x16B loads.
__device__ __forceinline__ v16h frag_a_f16(const _Float16* base, int ld) {
  int l = lane_id();
  const _Float16* row = base + (l & 15) * ld + (l >> 4) * 16;
  v8h lo = *(const v8h*)row;
  v8h hi = *(const v8h*)(row + 8);
  return cat8(lo, hi);
}

// B fragment from pre-swizzled "fragment dump" layout: tile = [32 lanes][16
// halfs] contiguous (element j of lane = B[k = 16*(lane>>4) + j][n = lane&15]).
__device__ __forceinline__ v16h frag_b_swz(const _Float16* tileBase) {
  const v8h* p = (const v8h*)(tileBase + lane_id() * 16);
  return cat8(p[0], p[1]);
}

// A fragment from the f32 edge-length-embedding, K padded 16->32 with zeros.
__device__ __forceinline__ v16h frag_a_elemb(const float* elemb, int e0) {
  int l = lane_id();
  int m = l & 15, half = l >> 4;
  const float* row = elemb + (size_t)(e0 + m) * NBAS;
  v16h a;
#pragma unroll
  for (int v = 0; v < 8; ++v) {
    int k = half * 16 + 2 * v;
    a[2 * v]     = (k     < NBAS) ? (_Float16)row[k]     : (_Float16)0.f;
    a[2 * v + 1] = (k + 1 < NBAS) ? (_Float16)row[k + 1] : (_Float16)0.f;
  }
  return a;
}

// ---------------- prep kernels ----------------
// bswz: per layer, B = [Wx | Wsc_a0..a3] (128 x 640) stored swizzled:
// [kk(4)][ntile(40)][lane(32)][j(16)]; element = B[kk*32+16*(lane>>4)+j][ntile*16+(lane&15)]
__global__ void prep_bcat_kernel(const float* __restrict__ Wx,
                                 const float* __restrict__ Wsc,
                                 _Float16* __restrict__ bswz) {
  int idx = blockIdx.x * blockDim.x + threadIdx.x;
  const int perL = 4 * 40 * 32 * 16;  // 81920
  const int total = NLAY * perL;
  if (idx >= total) return;
  int l = idx / perL;
  int rem = idx % perL;
  int kk = rem / (40 * 512);
  int rem2 = rem % (40 * 512);
  int ntile = rem2 / 512;
  int lane = (rem2 >> 4) & 31;
  int j = rem2 & 15;
  int k = kk * 32 + ((lane >> 4) << 4) + j;       // B row (0..127)
  int c = ntile * 16 + (lane & 15);               // B col (0..639)
  float v;
  if (c < DMUL) {
    v = Wx[((size_t)l * DMUL + k) * DMUL + c];
  } else {
    int cc = c - DMUL;
    int a = cc >> 7, e = cc & 127;
    v = Wsc[(((size_t)l * DMUL + k) * NATTR + a) * DMUL + e];
  }
  bswz[idx] = (_Float16)v;
}

// w21f[l] = (W2[l] @ Wp[l]) in f16, [256 x 128] row-major (fold, no nonlinearity)
__global__ void w21_kernel(const float* __restrict__ W2,
                           const float* __restrict__ Wp,
                           _Float16* __restrict__ w21f) {
  int idx = blockIdx.x * blockDim.x + threadIdx.x;
  const int total = NLAY * 256 * DMUL;
  if (idx >= total) return;
  int l = idx / (256 * DMUL);
  int i = (idx / DMUL) % 256;
  int j = idx % DMUL;
  const float* w2 = W2 + ((size_t)l * 256 + i) * 256;
  const float* wp = Wp + (size_t)l * 256 * DMUL + j;
  float s = 0.f;
  for (int k = 0; k < 256; ++k) s += w2[k] * wp[(size_t)k * DMUL];
  w21f[idx] = (_Float16)s;
}

__global__ void init_h_kernel(const float* __restrict__ x, float* __restrict__ h32,
                              _Float16* __restrict__ h16, float* __restrict__ newh) {
  size_t i = (size_t)blockIdx.x * blockDim.x + threadIdx.x;
  if (i >= (size_t)NNODES * DMUL) return;
  float v = x[i];
  h32[i] = v;
  h16[i] = (_Float16)v;
  newh[i] = 0.f;
}

// ---------------- node GEMM: [hx | sc] = h16 @ [Wx | Wsc_a0..a3] ----------------
// Register-blocked over N: each wave owns one mtile and a group of 8 ntiles
// (group 0 = hx columns; group g>=1 = the Wsc slice for a = g-1), keeping its
// 4 A fragments resident. The hx/sc split and the node_attr scale index are
// wave-uniform, so scales are preloaded once and the sc epilogue is pure
// mul + global_atomic_add_f32.
__global__ void node_gemm_kernel(const _Float16* __restrict__ h16,
                                 const _Float16* __restrict__ bswz_l,
                                 const float* __restrict__ node_attr,
                                 float* __restrict__ hx,
                                 float* __restrict__ newh) {
  const int NGRP = 5;               // 40 ntiles / 8 per wave
  const int MT = NNODES / 16;
  int wave = blockIdx.x * (blockDim.x >> 5) + (threadIdx.x >> 5);
  if (wave >= MT * NGRP) return;
  int mtile = wave / NGRP, ngrp = wave % NGRP;

  v16h a[4];
#pragma unroll
  for (int kk = 0; kk < 4; ++kk)
    a[kk] = frag_a_f16(h16 + (size_t)mtile * 16 * DMUL + kk * 32, DMUL);

  int l = lane_id();
  int n = l & 15, half = l >> 4;

  if (ngrp == 0) {
    // hx block: ntiles 0..7, plain stores
#pragma unroll
    for (int t = 0; t < 8; ++t) {
      v8f c = {};
#pragma unroll
      for (int kk = 0; kk < 4; ++kk)
        c = wmma16x16x32(a[kk], frag_b_swz(bswz_l + ((size_t)kk * 40 + t) * 512), c);
#pragma unroll
      for (int r = 0; r < 8; ++r) {
        int node = mtile * 16 + r + 8 * half;
        hx[(size_t)node * DMUL + t * 16 + n] = c[r];
      }
    }
  } else {
    // self-connection block for a = ngrp-1: preload wave-uniform-index scales
    int aidx = ngrp - 1;
    float scl[8];
#pragma unroll
    for (int r = 0; r < 8; ++r) {
      int node = mtile * 16 + r + 8 * half;
      scl[r] = node_attr[(size_t)node * NATTR + aidx];
    }
#pragma unroll
    for (int t = 0; t < 8; ++t) {
      int ntile = ngrp * 8 + t;
      v8f c = {};
#pragma unroll
      for (int kk = 0; kk < 4; ++kk)
        c = wmma16x16x32(a[kk], frag_b_swz(bswz_l + ((size_t)kk * 40 + ntile) * 512), c);
      int ecol = t * 16 + n;
#pragma unroll
      for (int r = 0; r < 8; ++r) {
        int node = mtile * 16 + r + 8 * half;
        atomicAdd(&newh[(size_t)node * DMUL + ecol], scl[r] * c[r]);
      }
    }
  }
}

// ---------------- fused edge kernel ----------------
// Radial MLP as 3 chained WMMA GEMMs per 16-edge tile; weights cached in LDS
// in fragment-dump (swizzled) layout so every B load is 2x ds_load_b128.
__global__ void edge_kernel(const float* __restrict__ elemb,
                            const float* __restrict__ edge_attr,
                            const int* __restrict__ edge_src,
                            const int* __restrict__ edge_dst,
                            const float* __restrict__ W0l,      // [16][128] f32
                            const float* __restrict__ W1l,      // [128][256] f32
                            const _Float16* __restrict__ W21l,  // [256][128] f16
                            const float* __restrict__ Wal,      // [9][128] f32
                            const float* __restrict__ hx,
                            float* __restrict__ newh) {
  extern __shared__ _Float16 smem[];
  _Float16* sW0  = smem;               // 8 tiles  x 512  (K=32 zero-padded)
  _Float16* sW1  = sW0 + 8 * 512;      // 64 tiles x 512  (kt*16 + nt)
  _Float16* sW21 = sW1 + 64 * 512;     // 64 tiles x 512  (kt*8 + nt)
  _Float16* sStg = sW21 + 64 * 512;    // per-wave: 16*128 + 16*256

  int tid = threadIdx.x;
  // fill LDS weights, swizzling into fragment-dump layout
  for (int i = tid; i < 8 * 512; i += blockDim.x) {
    int nt = i >> 9, lane = (i >> 4) & 31, j = i & 15;
    int k = ((lane >> 4) << 4) + j;
    int col = nt * 16 + (lane & 15);
    sW0[i] = (k < NBAS) ? (_Float16)W0l[k * DMUL + col] : (_Float16)0.f;
  }
  for (int i = tid; i < 64 * 512; i += blockDim.x) {
    int tile = i >> 9, lane = (i >> 4) & 31, j = i & 15;
    int kt = tile >> 4, nt = tile & 15;
    int k = kt * 32 + ((lane >> 4) << 4) + j;
    int col = nt * 16 + (lane & 15);
    sW1[i] = (_Float16)W1l[k * 256 + col];
  }
  for (int i = tid; i < 64 * 512; i += blockDim.x) {
    int tile = i >> 9, lane = (i >> 4) & 31, j = i & 15;
    int kt = tile >> 3, nt = tile & 7;
    int k = kt * 32 + ((lane >> 4) << 4) + j;
    int col = nt * 16 + (lane & 15);
    sW21[i] = W21l[k * DMUL + col];
  }
  __syncthreads();

  int wave = tid >> 5;
  _Float16* t0 = sStg + wave * (16 * DMUL + 16 * 256);  // [16][128]
  _Float16* t1 = t0 + 16 * DMUL;                        // [16][256]
  int lane = tid & 31;
  int n = lane & 15, half = lane >> 4;

  const int ET = NEDGES / 16;
  int gw = blockIdx.x * (blockDim.x >> 5) + wave;
  int gstride = gridDim.x * (blockDim.x >> 5);

  for (int tile = gw; tile < ET; tile += gstride) {
    int e0 = tile * 16;

    // ---- GEMM1: t0 = gelu(elemb_tile @ W0)  [16x128], K=16 (padded)
    v16h ael = frag_a_elemb(elemb, e0);
#pragma unroll
    for (int j = 0; j < 8; ++j) {
      v8f c = {};
      c = wmma16x16x32(ael, frag_b_swz(sW0 + j * 512), c);
#pragma unroll
      for (int r = 0; r < 8; ++r) {
        int m = r + 8 * half;
        t0[m * DMUL + j * 16 + n] = (_Float16)gelu_f(c[r]);
      }
    }

    // ---- GEMM2: t1 = gelu(t0 @ W1)  [16x256], K=128; A frags hoisted
    {
      v16h a2[4];
#pragma unroll
      for (int kk = 0; kk < 4; ++kk) a2[kk] = frag_a_f16(t0 + kk * 32, DMUL);
      for (int j = 0; j < 16; ++j) {
        v8f c = {};
#pragma unroll
        for (int kk = 0; kk < 4; ++kk)
          c = wmma16x16x32(a2[kk], frag_b_swz(sW1 + (kk * 16 + j) * 512), c);
#pragma unroll
        for (int r = 0; r < 8; ++r) {
          int m = r + 8 * half;
          t1[m * 256 + j * 16 + n] = (_Float16)gelu_f(c[r]);
        }
      }
    }

    // per-lane edge metadata (each lane covers 8 edge rows m = 8*half + r)
    int esrc[8], edst[8];
    float eatt[8][SHD];
#pragma unroll
    for (int r = 0; r < 8; ++r) {
      int e = e0 + r + 8 * half;
      esrc[r] = edge_src[e];
      edst[r] = edge_dst[e];
#pragma unroll
      for (int s = 0; s < SHD; ++s) eatt[r][s] = edge_attr[(size_t)e * SHD + s];
    }

    // ---- GEMM3: we = t1 @ (W2@Wp)  [16x128], K=256; A frags hoisted
    {
      v16h a3[8];
#pragma unroll
      for (int kk = 0; kk < 8; ++kk) a3[kk] = frag_a_f16(t1 + kk * 32, 256);
      for (int j = 0; j < 8; ++j) {
        v8f c = {};
#pragma unroll
        for (int kk = 0; kk < 8; ++kk)
          c = wmma16x16x32(a3[kk], frag_b_swz(sW21 + (kk * 8 + j) * 512), c);
        int col = j * 16 + n;
        float wa[SHD];
#pragma unroll
        for (int s = 0; s < SHD; ++s) wa[s] = Wal[s * DMUL + col];
#pragma unroll
        for (int r = 0; r < 8; ++r) {
          float ea = 0.f;
#pragma unroll
          for (int s = 0; s < SHD; ++s) ea += eatt[r][s] * wa[s];
          float hv = hx[(size_t)esrc[r] * DMUL + col];
          float msg = hv * ea * c[r] * INV_SQRT_Z;
          atomicAdd(&newh[(size_t)edst[r] * DMUL + col], msg);
        }
      }
    }
  }
}

// ---------------- layer epilogue ----------------
__global__ void finalize_kernel(float* __restrict__ h32, _Float16* __restrict__ h16,
                                float* __restrict__ newh, int apply_gelu) {
  size_t i = (size_t)blockIdx.x * blockDim.x + threadIdx.x;
  if (i >= (size_t)NNODES * DMUL) return;
  float v = newh[i];
  if (apply_gelu) v = gelu_f(v);
  h32[i] = v;
  h16[i] = (_Float16)v;
  newh[i] = 0.f;  // ready for next layer's sc/agg accumulation
}

// ---------------- pooling + head ----------------
__global__ void zero_g_kernel(float* __restrict__ g) {
  int i = blockIdx.x * blockDim.x + threadIdx.x;
  if (i < NGR * DMUL) g[i] = 0.f;
}

__global__ void pool_kernel(const float* __restrict__ h32, const int* __restrict__ batch,
                            float* __restrict__ g) {
  size_t i = (size_t)blockIdx.x * blockDim.x + threadIdx.x;
  if (i >= (size_t)NNODES * DMUL) return;
  int node = (int)(i / DMUL), col = (int)(i % DMUL);
  atomicAdd(&g[batch[node] * DMUL + col], h32[i]);
}

__global__ void head_kernel(const float* __restrict__ g,
                            const float* __restrict__ fc1w, const float* __restrict__ fc1b,
                            const float* __restrict__ fc2w, const float* __restrict__ fc2b,
                            const float* __restrict__ outw, const float* __restrict__ outb,
                            float* __restrict__ out) {
  __shared__ float sg[NGR * DMUL];
  __shared__ float s1[NGR * 4 * DMUL];
  __shared__ float s2[NGR * 4 * DMUL];
  __shared__ float slog[NGR * NDIH];
  int tid = threadIdx.x;
  for (int i = tid; i < NGR * DMUL; i += blockDim.x) sg[i] = g[i];
  __syncthreads();
  for (int o = tid; o < NGR * 512; o += blockDim.x) {
    int row = o >> 9, col = o & 511;
    float s = fc1b[col];
    for (int k = 0; k < DMUL; ++k) s += sg[row * DMUL + k] * fc1w[k * 512 + col];
    s1[o] = gelu_f(s);
  }
  __syncthreads();
  for (int o = tid; o < NGR * 512; o += blockDim.x) {
    int row = o >> 9, col = o & 511;
    float s = fc2b[col];
    for (int k = 0; k < 512; ++k) s += s1[row * 512 + k] * fc2w[k * 512 + col];
    s2[o] = gelu_f(s);
  }
  __syncthreads();
  for (int o = tid; o < NGR * NDIH; o += blockDim.x) {
    int row = o / NDIH, col = o % NDIH;
    float s = outb[col];
    for (int k = 0; k < 512; ++k) s += s2[row * 512 + k] * outw[k * NDIH + col];
    slog[o] = s;
  }
  __syncthreads();
  if (tid < NGR) {
    float mx = -1e30f;
    for (int c = 0; c < NDIH; ++c) mx = fmaxf(mx, slog[tid * NDIH + c]);
    float sum = 0.f, e[NDIH];
    for (int c = 0; c < NDIH; ++c) { e[c] = expf(slog[tid * NDIH + c] - mx); sum += e[c]; }
    for (int c = 0; c < NDIH; ++c) out[tid * NDIH + c] = e[c] / sum;
  }
}

// ---------------- launch ----------------
extern "C" void kernel_launch(void* const* d_in, const int* in_sizes, int n_in,
                              void* d_out, int out_size, void* d_ws, size_t ws_size,
                              hipStream_t stream) {
  const float* x     = (const float*)d_in[0];
  const float* nattr = (const float*)d_in[1];
  const float* eattr = (const float*)d_in[2];
  const float* elemb = (const float*)d_in[3];
  const int*   esrc  = (const int*)d_in[4];
  const int*   edst  = (const int*)d_in[5];
  const int*   batch = (const int*)d_in[6];
  const float* W0    = (const float*)d_in[7];
  const float* W1    = (const float*)d_in[8];
  const float* W2    = (const float*)d_in[9];
  const float* Wp    = (const float*)d_in[10];
  const float* Wx    = (const float*)d_in[11];
  const float* Wa    = (const float*)d_in[12];
  const float* Wsc   = (const float*)d_in[13];
  const float* fc1w  = (const float*)d_in[14];
  const float* fc1b  = (const float*)d_in[15];
  const float* fc2w  = (const float*)d_in[16];
  const float* fc2b  = (const float*)d_in[17];
  const float* outw  = (const float*)d_in[18];
  const float* outb  = (const float*)d_in[19];
  float* out = (float*)d_out;
  (void)in_sizes; (void)n_in; (void)out_size; (void)ws_size;

  // workspace carve-up (~55 MB total)
  char* base = (char*)d_ws;
  size_t off = 0;
  auto take = [&](size_t bytes) -> char* {
    char* p = base + off;
    off = (off + bytes + 255) & ~(size_t)255;
    return p;
  };
  float*    h32  = (float*)take((size_t)NNODES * DMUL * 4);
  _Float16* h16  = (_Float16*)take((size_t)NNODES * DMUL * 2);
  float*    hx   = (float*)take((size_t)NNODES * DMUL * 4);
  float*    newh = (float*)take((size_t)NNODES * DMUL * 4);
  _Float16* bswz = (_Float16*)take((size_t)NLAY * DMUL * 640 * 2);
  _Float16* w21f = (_Float16*)take((size_t)NLAY * 256 * DMUL * 2);
  float*    gbuf = (float*)take((size_t)NGR * DMUL * 4);

  const int TB = 256;
  const int nd_blocks = (NNODES * DMUL + TB - 1) / TB;

  prep_bcat_kernel<<<(NLAY * DMUL * 640 + TB - 1) / TB, TB, 0, stream>>>(Wx, Wsc, bswz);
  w21_kernel<<<(NLAY * 256 * DMUL + TB - 1) / TB, TB, 0, stream>>>(W2, Wp, w21f);
  init_h_kernel<<<nd_blocks, TB, 0, stream>>>(x, h32, h16, newh);

  // edge kernel LDS: swizzled weights (8+64+64 tiles x 1KB) + 8 waves staging
  const size_t edge_shmem =
      (size_t)(8 * 512 + 64 * 512 + 64 * 512 + 8 * (16 * DMUL + 16 * 256)) *
      sizeof(_Float16);  // ~232 KB, within 320 KB WGP budget

  const int node_waves = (NNODES / 16) * 5;  // 1875 mtiles x 5 ntile-groups
  const int ng_blocks = (node_waves + 7) / 8;

  for (int l = 0; l < NLAY; ++l) {
    node_gemm_kernel<<<ng_blocks, TB, 0, stream>>>(
        h16, bswz + (size_t)l * DMUL * 640, nattr, hx, newh);
    edge_kernel<<<512, TB, edge_shmem, stream>>>(
        elemb, eattr, esrc, edst,
        W0 + (size_t)l * NBAS * DMUL,
        W1 + (size_t)l * DMUL * 256,
        w21f + (size_t)l * 256 * DMUL,
        Wa + (size_t)l * SHD * DMUL,
        hx, newh);
    finalize_kernel<<<nd_blocks, TB, 0, stream>>>(h32, h16, newh, l < NLAY - 1 ? 1 : 0);
  }

  zero_g_kernel<<<(NGR * DMUL + TB - 1) / TB, TB, 0, stream>>>(gbuf);
  pool_kernel<<<nd_blocks, TB, 0, stream>>>(h32, batch, gbuf);
  head_kernel<<<1, TB, 0, stream>>>(gbuf, fc1w, fc1b, fc2w, fc2b, outw, outb, out);
}
